// VisionMamba_9517647528497
// MI455X (gfx1250) — matrix operand order repaired and checked by reading, not compiled, not run on hardware
//
#include <hip/hip_runtime.h>

// ---------------- model dims ----------------
#define B_SZ     16
#define DEPTH    24
#define D_MODEL  192
#define D_INNER  384
#define D_STATE  16
#define D_CONV   4
#define DT_RANK  12
#define L_TOK    320           // 64 z tokens + 256 x tokens
#define NTOK     (B_SZ * L_TOK)   // 5120
#define EPSV     1e-5f
#define XPROJ_N  (DT_RANK + 2 * D_STATE)   // 44
#define XPROJ_NP 48                         // padded to multiple of 16

// LDS B-panel row pitch in ushorts (80 bytes: 16B aligned, bank-spread)
#define BPITCH   40

// ---------------- workspace layout (bytes) ----------------
#define OFF_RESID   0u
#define OFF_HIDDEN  (OFF_RESID  + (unsigned)(NTOK*D_MODEL*4))        // fp32
#define OFF_HNBF    (OFF_HIDDEN + (unsigned)(NTOK*D_MODEL*4))        // bf16
#define OFF_XZ      (OFF_HNBF   + (unsigned)(NTOK*D_MODEL*2))        // fp32 (aliases im2col patches bf16)
#define OFF_XC      (OFF_XZ     + (unsigned)(NTOK*2*D_INNER*4))      // fp32
#define OFF_XCBF    (OFF_XC     + (unsigned)(NTOK*D_INNER*4))        // bf16
#define OFF_DBL     (OFF_XCBF   + (unsigned)(NTOK*D_INNER*2))        // fp32
#define OFF_YBF     (OFF_DBL    + (unsigned)(NTOK*XPROJ_NP*4))       // bf16
#define OFF_WIN     (OFF_YBF    + (unsigned)(NTOK*D_INNER*2))        // bf16
#define OFF_WXP     (OFF_WIN    + (unsigned)(DEPTH*2*D_INNER*D_MODEL*2))
#define OFF_WOUT    (OFF_WXP    + (unsigned)(DEPTH*XPROJ_NP*D_INNER*2))
#define OFF_WPT     (OFF_WOUT   + (unsigned)(DEPTH*D_MODEL*D_INNER*2))

typedef __attribute__((ext_vector_type(16))) __bf16 v16bf;
typedef __attribute__((ext_vector_type(8)))  float  v8f;

union FragU { v16bf v; uint4 q[2]; };

__device__ __forceinline__ unsigned short f2bf(float f) {
    unsigned int u = __float_as_uint(f);
    u += 0x7FFFu + ((u >> 16) & 1u);          // round to nearest even
    return (unsigned short)(u >> 16);
}

// ---------------- CDNA5 async global->LDS (ASYNCcnt-tracked) ----------------
// GLOBAL_LOAD_ASYNC_TO_LDS_B128: VDST = LDS byte address VGPR, VADDR = 64b addr.
__device__ __forceinline__ void async_g2l_b128(void* lds_ptr, const void* gptr) {
    unsigned lds_off = (unsigned)(uintptr_t)lds_ptr;     // low 32b of generic = LDS offset
    unsigned long long ga = (unsigned long long)(uintptr_t)gptr;
    asm volatile("global_load_async_to_lds_b128 %0, %1, off"
                 :: "v"(lds_off), "v"(ga) : "memory");
}
__device__ __forceinline__ void wait_async_le1() {
    asm volatile("s_wait_asynccnt 0x1" ::: "memory");
}
__device__ __forceinline__ void wait_async_le0() {
    asm volatile("s_wait_asynccnt 0x0" ::: "memory");
}

// ---------------- weight conversion ----------------
__global__ void cvt_bf16_k(unsigned short* __restrict__ dst,
                           const float* __restrict__ src, int n) {
    int i = blockIdx.x * blockDim.x + threadIdx.x;
    if (i < n) dst[i] = f2bf(src[i]);
}

// x_proj: (DEPTH,44,384) -> (DEPTH,48,384) bf16, zero-padded rows
__global__ void cvt_xproj_pad_k(unsigned short* __restrict__ dst,
                                const float* __restrict__ src) {
    int i = blockIdx.x * blockDim.x + threadIdx.x;
    int total = DEPTH * XPROJ_NP * D_INNER;
    if (i >= total) return;
    int k = i % D_INNER;
    int n = (i / D_INNER) % XPROJ_NP;
    int layer = i / (D_INNER * XPROJ_NP);
    float v = (n < XPROJ_N) ? src[(layer * XPROJ_N + n) * D_INNER + k] : 0.0f;
    dst[i] = f2bf(v);
}

// ---------------- im2col for patch embed ----------------
__global__ void im2col_k(const float* __restrict__ xi, const float* __restrict__ zi,
                         unsigned short* __restrict__ patches) {
    int idx = blockIdx.x * blockDim.x + threadIdx.x;
    if (idx >= NTOK * 768) return;
    int k = idx % 768;
    int t = idx / 768;
    int l = t % L_TOK;
    int b = t / L_TOK;
    int c = k / 256;
    int r = k % 256;
    int ii = r / 16, jj = r % 16;
    float v;
    if (l < 64) {
        int ph = l / 8, pw = l % 8;
        v = zi[((b * 3 + c) * 128 + ph * 16 + ii) * 128 + pw * 16 + jj];
    } else {
        int lx = l - 64;
        int ph = lx / 16, pw = lx % 16;
        v = xi[((b * 3 + c) * 256 + ph * 16 + ii) * 256 + pw * 16 + jj];
    }
    patches[idx] = f2bf(v);
}

__global__ void add_pos_k(float* __restrict__ hidden,
                          const float* __restrict__ pos_x,
                          const float* __restrict__ pos_z) {
    int idx = blockIdx.x * blockDim.x + threadIdx.x;
    if (idx >= NTOK * D_MODEL) return;
    int d = idx % D_MODEL;
    int t = idx / D_MODEL;
    int l = t % L_TOK;
    float p = (l < 64) ? pos_z[l * D_MODEL + d] : pos_x[(l - 64) * D_MODEL + d];
    hidden[idx] += p;
}

// ---------------- WMMA fragment loaders (ISA 7.12.2 layouts) ----------------
// A 16x32 bf16: lane m=lane&15, half h=lane>>4; two contiguous 8-elem (16B) runs.
__device__ __forceinline__ v16bf load_frag_a(const unsigned short* __restrict__ A,
                                             int lda, int m0, int k0) {
    int lane = threadIdx.x & 31;
    int m = m0 + (lane & 15);
    int h = lane >> 4;
    const unsigned short* p = A + (size_t)m * lda + k0 + h * 8;
    FragU u;
    u.q[0] = *(const uint4*)(p);
    u.q[1] = *(const uint4*)(p + 16);
    return u.v;
}

// B 32x16 frag sourced from LDS panel: row n (ushort pitch BPITCH) holds K=0..31.
// lanes 0-15: K=0..7,16..23, lanes 16-31: K=8..15,24..31 -> two 16B runs per lane.
__device__ __forceinline__ v16bf load_frag_b_lds(const unsigned short* sB, int tile) {
    int lane = threadIdx.x & 31;
    int n = tile * 16 + (lane & 15);
    int h = lane >> 4;
    const unsigned short* p = sB + n * BPITCH + h * 8;
    FragU u;
    u.q[0] = *(const uint4*)(p);
    u.q[1] = *(const uint4*)(p + 16);
    return u.v;
}

// ---------------- LDS-staged register-blocked WMMA GEMM ----------------
// C(MxN) = A(MxK) @ W(NxK)^T (+bias).
// Block = 8 waves. Each wave: 2 M-tiles x NT N-tiles (32 rows x NT*16 cols).
// Block tile: 256 rows x NT*16 cols. B-panel (NT*16 x 32 per K-step) staged into
// LDS via async global->LDS b128, double-buffered, shared by all 8 waves.
template <int NT>
__global__ void gemm_lds_wmma_k(const unsigned short* __restrict__ A, int lda,
                                const unsigned short* __restrict__ W, int ldw,
                                float* __restrict__ C, int ldc,
                                int M, int K,
                                const float* __restrict__ bias) {
    __shared__ unsigned short sB[2][NT * 16 * BPITCH];

    const int tid  = threadIdx.x;
    const int wid  = tid >> 5;
    const int lane = tid & 31;
    const int n0   = blockIdx.x * (NT * 16);
    const int m0   = blockIdx.y * 256 + wid * 32;
    const int Ksteps = K >> 5;

    // cooperative async stage of one 32-deep B panel into buffer `buf`
    // thread t (t < NT*64): n = t>>2, 16B chunk = t&3
    auto stage = [&](int buf, int k0) {
        if (NT == 4 || tid < NT * 64) {
            int n = tid >> 2;
            int chunk = tid & 3;
            const unsigned short* g = W + (size_t)(n0 + n) * ldw + k0 + chunk * 8;
            unsigned short* l = &sB[buf][n * BPITCH + chunk * 8];
            async_g2l_b128(l, g);
        }
    };

    v8f acc[2][NT];
#pragma unroll
    for (int mt = 0; mt < 2; ++mt)
#pragma unroll
        for (int j = 0; j < NT; ++j)
            acc[mt][j] = (v8f){0.f, 0.f, 0.f, 0.f, 0.f, 0.f, 0.f, 0.f};

    stage(0, 0);
    for (int ks = 0; ks < Ksteps; ++ks) {
        if (ks + 1 < Ksteps) {
            stage((ks + 1) & 1, (ks + 1) << 5);
            wait_async_le1();                 // current panel complete
        } else {
            wait_async_le0();
        }
        __syncthreads();                      // panel visible to all waves

        v16bf a0 = load_frag_a(A, lda, m0,      ks << 5);
        v16bf a1 = load_frag_a(A, lda, m0 + 16, ks << 5);
        v16bf bf[NT];
#pragma unroll
        for (int j = 0; j < NT; ++j)          // issue all LDS reads first
            bf[j] = load_frag_b_lds(sB[ks & 1], j);
#pragma unroll
        for (int j = 0; j < NT; ++j)          // then dense WMMA bursts
            acc[0][j] = __builtin_amdgcn_wmma_f32_16x16x32_bf16(
                false, a0, false, bf[j], (short)0, acc[0][j], false, false);
#pragma unroll
        for (int j = 0; j < NT; ++j)
            acc[1][j] = __builtin_amdgcn_wmma_f32_16x16x32_bf16(
                false, a1, false, bf[j], (short)0, acc[1][j], false, false);

        __syncthreads();                      // readers done before next overwrite
    }

    int nlo = lane & 15;
#pragma unroll
    for (int mt = 0; mt < 2; ++mt) {
        int mbase = m0 + mt * 16 + ((lane >> 4) << 3);
#pragma unroll
        for (int j = 0; j < NT; ++j) {
            int n = n0 + j * 16 + nlo;
            float bs = bias ? bias[n] : 0.0f;
#pragma unroll
            for (int v = 0; v < 8; ++v)
                C[(size_t)(mbase + v) * ldc + n] = acc[mt][j][v] + bs;
        }
    }
}

// ---------------- fused add-residual + LayerNorm -> bf16 ----------------
__global__ void addnorm_k(const float* __restrict__ hid, float* __restrict__ resid,
                          unsigned short* __restrict__ hnbf,
                          const float* __restrict__ w, const float* __restrict__ b,
                          int first) {
    int tok  = (int)((blockIdx.x * blockDim.x + threadIdx.x) >> 5);
    int lane = threadIdx.x & 31;
    if (tok >= NTOK) return;
    const float* hrow = hid + (size_t)tok * D_MODEL;
    float* rrow = resid + (size_t)tok * D_MODEL;

    float vals[6];
    float s = 0.f;
#pragma unroll
    for (int i = 0; i < 6; ++i) {
        int c = lane + i * 32;
        float v = hrow[c] + (first ? 0.f : rrow[c]);
        vals[i] = v;
        rrow[c] = v;
        s += v;
    }
#pragma unroll
    for (int o = 16; o > 0; o >>= 1) s += __shfl_xor(s, o, 32);
    float mu = s * (1.0f / D_MODEL);
    float sq = 0.f;
#pragma unroll
    for (int i = 0; i < 6; ++i) { float d = vals[i] - mu; sq += d * d; }
#pragma unroll
    for (int o = 16; o > 0; o >>= 1) sq += __shfl_xor(sq, o, 32);
    float rstd = rsqrtf(sq * (1.0f / D_MODEL) + EPSV);
#pragma unroll
    for (int i = 0; i < 6; ++i) {
        int c = lane + i * 32;
        hnbf[(size_t)tok * D_MODEL + c] = f2bf((vals[i] - mu) * rstd * w[c] + b[c]);
    }
}

// ---------------- causal depthwise conv1d + SiLU ----------------
__global__ void conv_silu_k(const float* __restrict__ xz,
                            const float* __restrict__ cw, const float* __restrict__ cb,
                            float* __restrict__ xc, unsigned short* __restrict__ xcbf) {
    int idx = blockIdx.x * blockDim.x + threadIdx.x;
    if (idx >= NTOK * D_INNER) return;
    int d = idx % D_INNER;
    int t = idx / D_INNER;
    int l = t % L_TOK;
    float acc = cb[d];
#pragma unroll
    for (int j = 0; j < D_CONV; ++j) {
        int ll = l - (D_CONV - 1) + j;
        if (ll >= 0)
            acc += cw[d * D_CONV + j] * xz[(size_t)(t - (D_CONV - 1 - j)) * (2 * D_INNER) + d];
    }
    float s = acc / (1.0f + __expf(-acc));
    xc[idx] = s;
    xcbf[idx] = f2bf(s);
}

// ---------------- fused dt_proj + softplus + selective scan + gate ----------------
__global__ void scan_k(const float* __restrict__ dbl,      // NTOK x 48 (dt|B|C)
                       const float* __restrict__ xz,       // NTOK x 768 (gate half)
                       const float* __restrict__ xc,       // NTOK x 384
                       const float* __restrict__ dtw,      // 384 x 12
                       const float* __restrict__ dtb,      // 384
                       const float* __restrict__ Alog,     // 384 x 16
                       const float* __restrict__ Dskip,    // 384
                       unsigned short* __restrict__ ybf) { // NTOK x 384 bf16
    int id = blockIdx.x * blockDim.x + threadIdx.x;
    if (id >= B_SZ * D_INNER) return;
    int b = id / D_INNER;
    int d = id % D_INNER;

    float wdt[DT_RANK];
#pragma unroll
    for (int j = 0; j < DT_RANK; ++j) wdt[j] = dtw[d * DT_RANK + j];
    float Arow[D_STATE];
#pragma unroll
    for (int n = 0; n < D_STATE; ++n) Arow[n] = -__expf(Alog[d * D_STATE + n]);
    float h[D_STATE];
#pragma unroll
    for (int n = 0; n < D_STATE; ++n) h[n] = 0.f;
    const float bias = dtb[d];
    const float Dsk = Dskip[d];

    for (int l = 0; l < L_TOK; ++l) {
        int t = b * L_TOK + l;
        const float* row = dbl + (size_t)t * XPROJ_NP;
        if (l + 1 < L_TOK) __builtin_prefetch(dbl + (size_t)(t + 1) * XPROJ_NP, 0, 0);
        float dtv = bias;
#pragma unroll
        for (int j = 0; j < DT_RANK; ++j) dtv += row[j] * wdt[j];
        dtv = (dtv > 20.f) ? dtv : log1pf(__expf(dtv));        // softplus
        float x = xc[(size_t)t * D_INNER + d];
        float y = 0.f;
#pragma unroll
        for (int n = 0; n < D_STATE; ++n) {
            float dA = __expf(dtv * Arow[n]);
            h[n] = dA * h[n] + dtv * row[DT_RANK + n] * x;
            y += h[n] * row[DT_RANK + D_STATE + n];
        }
        y += x * Dsk;
        float zg = xz[(size_t)t * (2 * D_INNER) + D_INNER + d];
        y *= zg / (1.0f + __expf(-zg));                        // * silu(z)
        ybf[(size_t)t * D_INNER + d] = f2bf(y);
    }
}

// ---------------- final add + LayerNorm -> fp32 out ----------------
__global__ void final_norm_k(const float* __restrict__ hid, const float* __restrict__ resid,
                             const float* __restrict__ w, const float* __restrict__ b,
                             float* __restrict__ out) {
    int tok  = (int)((blockIdx.x * blockDim.x + threadIdx.x) >> 5);
    int lane = threadIdx.x & 31;
    if (tok >= NTOK) return;
    float vals[6];
    float s = 0.f;
#pragma unroll
    for (int i = 0; i < 6; ++i) {
        int c = lane + i * 32;
        float v = hid[(size_t)tok * D_MODEL + c] + resid[(size_t)tok * D_MODEL + c];
        vals[i] = v; s += v;
    }
#pragma unroll
    for (int o = 16; o > 0; o >>= 1) s += __shfl_xor(s, o, 32);
    float mu = s * (1.0f / D_MODEL);
    float sq = 0.f;
#pragma unroll
    for (int i = 0; i < 6; ++i) { float d = vals[i] - mu; sq += d * d; }
#pragma unroll
    for (int o = 16; o > 0; o >>= 1) sq += __shfl_xor(sq, o, 32);
    float rstd = rsqrtf(sq * (1.0f / D_MODEL) + EPSV);
#pragma unroll
    for (int i = 0; i < 6; ++i) {
        int c = lane + i * 32;
        out[(size_t)tok * D_MODEL + c] = (vals[i] - mu) * rstd * w[c] + b[c];
    }
}

// ---------------- host launcher ----------------
extern "C" void kernel_launch(void* const* d_in, const int* in_sizes, int n_in,
                              void* d_out, int out_size, void* d_ws, size_t ws_size,
                              hipStream_t stream) {
    const float* x_img   = (const float*)d_in[0];
    const float* z_img   = (const float*)d_in[1];
    const float* patch_w = (const float*)d_in[2];
    const float* patch_b = (const float*)d_in[3];
    const float* pos_x   = (const float*)d_in[4];
    const float* pos_z   = (const float*)d_in[5];
    const float* ln_w    = (const float*)d_in[6];
    const float* ln_b    = (const float*)d_in[7];
    const float* in_w    = (const float*)d_in[8];
    const float* conv_w  = (const float*)d_in[9];
    const float* conv_b  = (const float*)d_in[10];
    const float* xproj_w = (const float*)d_in[11];
    const float* dt_w    = (const float*)d_in[12];
    const float* dt_b    = (const float*)d_in[13];
    const float* A_log   = (const float*)d_in[14];
    const float* D_skip  = (const float*)d_in[15];
    const float* out_w   = (const float*)d_in[16];
    const float* fnorm_w = (const float*)d_in[17];
    const float* fnorm_b = (const float*)d_in[18];

    char* ws = (char*)d_ws;
    float*          resid   = (float*)(ws + OFF_RESID);
    float*          hidden  = (float*)(ws + OFF_HIDDEN);
    unsigned short* hnbf    = (unsigned short*)(ws + OFF_HNBF);
    float*          xz      = (float*)(ws + OFF_XZ);
    unsigned short* patches = (unsigned short*)(ws + OFF_XZ);   // alias (used before xz)
    float*          xc      = (float*)(ws + OFF_XC);
    unsigned short* xcbf    = (unsigned short*)(ws + OFF_XCBF);
    float*          dbl     = (float*)(ws + OFF_DBL);
    unsigned short* ybf     = (unsigned short*)(ws + OFF_YBF);
    unsigned short* w_in    = (unsigned short*)(ws + OFF_WIN);
    unsigned short* w_xp    = (unsigned short*)(ws + OFF_WXP);
    unsigned short* w_out   = (unsigned short*)(ws + OFF_WOUT);
    unsigned short* w_pt    = (unsigned short*)(ws + OFF_WPT);

    const int TPB = 256;
    const int MSTRIPS = NTOK / 256;   // 20 (block tile = 256 rows)

    // 1) weight conversion fp32 -> bf16
    {
        int n = DEPTH * 2 * D_INNER * D_MODEL;
        cvt_bf16_k<<<(n + TPB - 1) / TPB, TPB, 0, stream>>>(w_in, in_w, n);
        n = DEPTH * D_MODEL * D_INNER;
        cvt_bf16_k<<<(n + TPB - 1) / TPB, TPB, 0, stream>>>(w_out, out_w, n);
        n = D_MODEL * 768;
        cvt_bf16_k<<<(n + TPB - 1) / TPB, TPB, 0, stream>>>(w_pt, patch_w, n);
        n = DEPTH * XPROJ_NP * D_INNER;
        cvt_xproj_pad_k<<<(n + TPB - 1) / TPB, TPB, 0, stream>>>(w_xp, xproj_w);
    }

    // 2) patch embed: im2col -> WMMA GEMM (M=5120,N=192,K=768) -> + pos
    {
        int n = NTOK * 768;
        im2col_k<<<(n + TPB - 1) / TPB, TPB, 0, stream>>>(x_img, z_img, patches);
        gemm_lds_wmma_k<4><<<dim3(D_MODEL / 64, MSTRIPS), TPB, 0, stream>>>(
            patches, 768, w_pt, 768, hidden, D_MODEL, NTOK, 768, patch_b);
        n = NTOK * D_MODEL;
        add_pos_k<<<(n + TPB - 1) / TPB, TPB, 0, stream>>>(hidden, pos_x, pos_z);
    }

    // 3) 24 Mamba layers
    for (int i = 0; i < DEPTH; ++i) {
        addnorm_k<<<NTOK / 8, TPB, 0, stream>>>(
            hidden, resid, hnbf, ln_w + i * D_MODEL, ln_b + i * D_MODEL, i == 0);

        // in_proj: M=5120, N=768, K=192
        gemm_lds_wmma_k<4><<<dim3((2 * D_INNER) / 64, MSTRIPS), TPB, 0, stream>>>(
            hnbf, D_MODEL, w_in + (size_t)i * 2 * D_INNER * D_MODEL, D_MODEL,
            xz, 2 * D_INNER, NTOK, D_MODEL, nullptr);

        {   // causal depthwise conv + SiLU
            int n = NTOK * D_INNER;
            conv_silu_k<<<(n + TPB - 1) / TPB, TPB, 0, stream>>>(
                xz, conv_w + (size_t)i * D_INNER * D_CONV, conv_b + (size_t)i * D_INNER,
                xc, xcbf);
        }

        // x_proj: M=5120, N=48(pad), K=384  (3 N-tiles per wave -> full row)
        gemm_lds_wmma_k<3><<<dim3(1, MSTRIPS), TPB, 0, stream>>>(
            xcbf, D_INNER, w_xp + (size_t)i * XPROJ_NP * D_INNER, D_INNER,
            dbl, XPROJ_NP, NTOK, D_INNER, nullptr);

        // dt_proj + softplus + selective scan + D-skip + SiLU gate
        scan_k<<<(B_SZ * D_INNER) / TPB, TPB, 0, stream>>>(
            dbl, xz, xc,
            dt_w + (size_t)i * D_INNER * DT_RANK, dt_b + (size_t)i * D_INNER,
            A_log + (size_t)i * D_INNER * D_STATE, D_skip + (size_t)i * D_INNER, ybf);

        // out_proj: M=5120, N=192, K=384
        gemm_lds_wmma_k<4><<<dim3(D_MODEL / 64, MSTRIPS), TPB, 0, stream>>>(
            ybf, D_INNER, w_out + (size_t)i * D_MODEL * D_INNER, D_INNER,
            hidden, D_MODEL, NTOK, D_INNER, nullptr);
    }

    // 4) final add + LayerNorm
    final_norm_k<<<NTOK / 8, TPB, 0, stream>>>(hidden, resid, fnorm_w, fnorm_b, (float*)d_out);
}
